// vq_Attention_20959440404840
// MI455X (gfx1250) — matrix operand order repaired and checked by reading, not compiled, not run on hardware
//
#include <hip/hip_runtime.h>
#include <hip/hip_bf16.h>

#define BATCH 64
#define SEQ 197
#define CH 768
#define HEADS 12
#define HDIM 64
#define SEQ_PAD 208
#define DICN 1000
#define MROWS (BATCH * SEQ)   /* 12608 = 788*16 exactly */
#define NQKV (3 * CH)         /* 2304 */
#define MT (MROWS / 16)       /* 788 */
#define NT4 (NQKV / 64)       /* 36: each wave does a 16x64 tile */
#define QTILES 13             /* ceil(197/16) */
#define KBLOCKS 7             /* ceil(197/32) */

typedef __attribute__((ext_vector_type(16))) __bf16 v16bf;
typedef __attribute__((ext_vector_type(8)))  float  v8f;

union Frag16 { unsigned u[8]; v16bf v; };

#define HAVE_ASYNC (__has_builtin(__builtin_amdgcn_global_load_async_to_lds_b128))

#if __has_builtin(__builtin_amdgcn_s_wait_asynccnt)
#define WAIT_ASYNC_N(n) do { __builtin_amdgcn_s_wait_asynccnt(n); asm volatile("" ::: "memory"); } while (0)
#else
#define WAIT_ASYNC_N(n) asm volatile("s_wait_asynccnt %0" ::: "memory")
#endif

// async copy element/pointer types (builtin signature: (int4 AS1*, int4 AS3*, Ii, Ii))
typedef int async_v4i __attribute__((ext_vector_type(4)));
typedef __attribute__((address_space(1))) async_v4i as1_v4i;
typedef __attribute__((address_space(3))) async_v4i as3_v4i;

__device__ __forceinline__ unsigned short f32_bf16(float f) {
  unsigned u = __float_as_uint(f);
  u += 0x7FFFu + ((u >> 16) & 1u);   // round-to-nearest-even
  return (unsigned short)(u >> 16);
}
__device__ __forceinline__ unsigned pack_bf16(float a, float b) {
  return (unsigned)f32_bf16(a) | ((unsigned)f32_bf16(b) << 16);
}
// 16-bit A fragment (16x32): lane row = lane&15; VGPR j holds K pair base:
__device__ __forceinline__ int kA(int j, int half) {
  return ((j < 4) ? (2 * j) : (2 * j + 8)) + 8 * half;
}
// 16-bit B fragment (32x16): lane col = lane&15; VGPR j holds K pair base (same as A):
__device__ __forceinline__ int kB(int j, int half) {
  return ((j < 4) ? (2 * j) : (2 * j + 8)) + 8 * half;
}

// ---------------------------------------------------------------------------
// Zero padding rows of q/k/vT staging buffers + loss accumulator.
// ---------------------------------------------------------------------------
__global__ void init_pad_kernel(unsigned short* __restrict__ qb,
                                unsigned short* __restrict__ kb,
                                unsigned short* __restrict__ vtb,
                                float* __restrict__ loss_acc) {
  int i = blockIdx.x * blockDim.x + threadIdx.x;
  if (i == 0) *loss_acc = 0.0f;
  const int padN = BATCH * HEADS * (SEQ_PAD - SEQ) * HDIM;  // 540672
  if (i < padN) {
    int d = i & 63;
    int t = i >> 6;
    int row = SEQ + (t % (SEQ_PAD - SEQ));
    int bh = t / (SEQ_PAD - SEQ);
    qb[(bh * SEQ_PAD + row) * HDIM + d] = 0;
    kb[(bh * SEQ_PAD + row) * HDIM + d] = 0;
    vtb[(bh * HDIM + d) * SEQ_PAD + row] = 0;
  }
}

// ---------------------------------------------------------------------------
// QKV GEMM: qkv = x @ qkv_w.T via bf16 WMMA. One wave = one 16x64 output
// tile (4 accumulators share one A fragment -> 4x fewer A loads/converts).
// Scatter into q (pre-scaled bf16 [BH,208,64]), k (bf16 [BH,208,64]),
// vT (bf16 [BH,64,208]).
// ---------------------------------------------------------------------------
__global__ __launch_bounds__(128) void qkv_wmma_kernel(
    const float* __restrict__ x, const float* __restrict__ w,
    unsigned short* __restrict__ qb, unsigned short* __restrict__ kb,
    unsigned short* __restrict__ vtb) {
  const int wave = blockIdx.x * (blockDim.x >> 5) + (threadIdx.x >> 5);
  const int lane = threadIdx.x & 31;
  const int half = lane >> 4;
  const int l15 = lane & 15;
  const int mt = wave / NT4, nt = wave % NT4;
  const int m0 = mt * 16, n0 = nt * 64;
  const float* __restrict__ pa_base = x + (size_t)(m0 + l15) * CH;
  const float* __restrict__ pb_base[4];
#pragma unroll
  for (int t = 0; t < 4; ++t)
    pb_base[t] = w + (size_t)(n0 + t * 16 + l15) * CH;

  v8f c[4];
#pragma unroll
  for (int t = 0; t < 4; ++t) c[t] = (v8f){0.f, 0.f, 0.f, 0.f, 0.f, 0.f, 0.f, 0.f};

  for (int k0 = 0; k0 < CH; k0 += 32) {
    if (k0 + 32 < CH) {
      __builtin_prefetch(pa_base + k0 + 32 + 8 * half, 0, 1);
      __builtin_prefetch(pb_base[half] + k0 + 32, 0, 1);
      __builtin_prefetch(pb_base[2 + half] + k0 + 32, 0, 1);
    }
    Frag16 a;
#pragma unroll
    for (int j = 0; j < 8; ++j) {
      const float* pa = pa_base + k0 + kA(j, half);
      a.u[j] = pack_bf16(pa[0], pa[1]);
    }
#pragma unroll
    for (int t = 0; t < 4; ++t) {
      Frag16 b;
#pragma unroll
      for (int j = 0; j < 8; ++j) {
        const float* pb = pb_base[t] + k0 + kB(j, half);
        b.u[j] = pack_bf16(pb[0], pb[1]);
      }
      c[t] = __builtin_amdgcn_wmma_f32_16x16x32_bf16(false, a.v, false, b.v,
                                                     (short)0, c[t], false, false);
    }
  }
  // scatter (which/h uniform per 16-col subtile since 16 | 768 and 16 | 64)
#pragma unroll
  for (int t = 0; t < 4; ++t) {
#pragma unroll
    for (int r = 0; r < 8; ++r) {
      int m = m0 + r + 8 * half;
      int n = n0 + t * 16 + l15;
      float val = c[t][r];
      int which = n / CH;
      int hd = n % CH;
      int h = hd >> 6, d = hd & 63;
      int bi = m / SEQ, nn = m % SEQ;
      int bh = bi * HEADS + h;
      if (which == 0) {
        qb[(bh * SEQ_PAD + nn) * HDIM + d] = f32_bf16(val * 0.125f);
      } else if (which == 1) {
        kb[(bh * SEQ_PAD + nn) * HDIM + d] = f32_bf16(val);
      } else {
        vtb[(bh * HDIM + d) * SEQ_PAD + nn] = f32_bf16(val);
      }
    }
  }
}

// ---------------------------------------------------------------------------
// Flash attention: one wave = one (b,h, 16-row q tile). Online softmax over
// 7 key blocks of 32. K/V tiles staged global->LDS with async b128 copies,
// double-buffered (copy of block kb+1 overlaps WMMAs of block kb; async
// loads complete in order, so waiting asynccnt<=16 releases the older 16).
// P transposed C-layout -> A-layout through LDS (s_wait_dscnt).
// ---------------------------------------------------------------------------
__global__ __launch_bounds__(128) void attn_wmma_kernel(
    const unsigned short* __restrict__ q, const unsigned short* __restrict__ k,
    const unsigned short* __restrict__ vt, float* __restrict__ xo) {
#if HAVE_ASYNC
  __shared__ unsigned short kvlds[4][2][32 * 64 + 64 * 32];  // 64 KB
#endif
  __shared__ unsigned short plds[4][16 * 32];                // 4 KB
  const int wslot = threadIdx.x >> 5;
  const int wave = blockIdx.x * 4 + wslot;
  const int lane = threadIdx.x & 31;
  const int half = lane >> 4;
  const int l15 = lane & 15;
  const int bh = wave / QTILES;
  const int qt = wave % QTILES;
  const int q0 = qt * 16;
  const unsigned short* __restrict__ kraw = k + (size_t)bh * SEQ_PAD * HDIM;
  const unsigned short* __restrict__ vraw = vt + (size_t)bh * HDIM * SEQ_PAD;
  const unsigned* __restrict__ qbase = (const unsigned*)(q + (size_t)bh * SEQ_PAD * HDIM);

#if HAVE_ASYNC
  auto async_copy_block = [&](int key0, int slot) {
    as3_v4i* kd = (as3_v4i*)(void*)&kvlds[wslot][slot][0];
    as3_v4i* vd = kd + 256;  // K tile is 4 KB = 256 x 16 B
    as1_v4i* kg = (as1_v4i*)((const char*)kraw + (size_t)key0 * (HDIM * 2));
    as1_v4i* vg = (as1_v4i*)((const char*)vraw + (size_t)key0 * 2);
#pragma unroll
    for (int i = 0; i < 8; ++i) {            // K tile: contiguous 4 KB
      int idx = i * 32 + lane;
      __builtin_amdgcn_global_load_async_to_lds_b128(kg + idx, kd + idx, 0, 0);
    }
#pragma unroll
    for (int i = 0; i < 8; ++i) {            // vT tile: 64 rows x 64 B (row stride 416 B = 26 x 16 B)
      int idx = i * 32 + lane;
      int d = idx >> 2, seg = idx & 3;
      __builtin_amdgcn_global_load_async_to_lds_b128(vg + (d * 26 + seg), vd + idx, 0, 0);
    }
  };
#endif

  Frag16 aq0, aq1;
#pragma unroll
  for (int j = 0; j < 8; ++j) {
    int kk = kA(j, half);
    aq0.u[j] = qbase[((q0 + l15) * HDIM + kk) >> 1];
    aq1.u[j] = qbase[((q0 + l15) * HDIM + 32 + kk) >> 1];
  }

  float mrow[8], lrow[8];
  v8f o[4];
#pragma unroll
  for (int r = 0; r < 8; ++r) { mrow[r] = -1e30f; lrow[r] = 0.f; }
#pragma unroll
  for (int t = 0; t < 4; ++t) o[t] = (v8f){0.f, 0.f, 0.f, 0.f, 0.f, 0.f, 0.f, 0.f};

#if HAVE_ASYNC
  async_copy_block(0, 0);
#endif

  for (int kb7 = 0; kb7 < KBLOCKS; ++kb7) {
    const int key0 = kb7 * 32;
#if HAVE_ASYNC
    if (kb7 + 1 < KBLOCKS) {
      async_copy_block(key0 + 32, (kb7 + 1) & 1);
      WAIT_ASYNC_N(16);
    } else {
      WAIT_ASYNC_N(0);
    }
    const unsigned* kl32 = (const unsigned*)&kvlds[wslot][kb7 & 1][0];
    const unsigned* vl32 = (const unsigned*)&kvlds[wslot][kb7 & 1][32 * 64];
#endif
    v8f s[2];
#pragma unroll
    for (int sub = 0; sub < 2; ++sub) {
      Frag16 bk0, bk1;
#pragma unroll
      for (int j = 0; j < 8; ++j) {
        int kk = kB(j, half);
#if HAVE_ASYNC
        int krow = sub * 16 + l15;                       // local row in LDS tile
        bk0.u[j] = kl32[(krow * HDIM + kk) >> 1];
        bk1.u[j] = kl32[(krow * HDIM + 32 + kk) >> 1];
#else
        int krow = key0 + sub * 16 + l15;
        bk0.u[j] = ((const unsigned*)kraw)[(krow * HDIM + kk) >> 1];
        bk1.u[j] = ((const unsigned*)kraw)[(krow * HDIM + 32 + kk) >> 1];
#endif
      }
      v8f cs = {0.f, 0.f, 0.f, 0.f, 0.f, 0.f, 0.f, 0.f};
      cs = __builtin_amdgcn_wmma_f32_16x16x32_bf16(false, aq0.v, false, bk0.v,
                                                   (short)0, cs, false, false);
      cs = __builtin_amdgcn_wmma_f32_16x16x32_bf16(false, aq1.v, false, bk1.v,
                                                   (short)0, cs, false, false);
      if (key0 + sub * 16 + l15 >= SEQ) {
#pragma unroll
        for (int r = 0; r < 8; ++r) cs[r] = -1e30f;
      }
      s[sub] = cs;
    }
    // online softmax row stats (xor masks 1..8 stay inside each 16-lane half)
    float alpha[8];
#pragma unroll
    for (int r = 0; r < 8; ++r) {
      float t = fmaxf(s[0][r], s[1][r]);
#pragma unroll
      for (int off = 8; off >= 1; off >>= 1) t = fmaxf(t, __shfl_xor(t, off, 32));
      float mnew = fmaxf(mrow[r], t);
      alpha[r] = __expf(mrow[r] - mnew);
      mrow[r] = mnew;
    }
#pragma unroll
    for (int r = 0; r < 8; ++r) {
      float p0 = __expf(s[0][r] - mrow[r]);
      float p1 = __expf(s[1][r] - mrow[r]);
      s[0][r] = p0; s[1][r] = p1;
      float t = p0 + p1;
#pragma unroll
      for (int off = 8; off >= 1; off >>= 1) t += __shfl_xor(t, off, 32);
      lrow[r] = lrow[r] * alpha[r] + t;
      o[0][r] *= alpha[r]; o[1][r] *= alpha[r];
      o[2][r] *= alpha[r]; o[3][r] *= alpha[r];
    }
    // P: C-layout -> LDS -> A-layout (bf16)
    unsigned short* pl = plds[wslot];
#pragma unroll
    for (int r = 0; r < 8; ++r) {
      int row = r + 8 * half;
      pl[row * 32 + l15] = f32_bf16(s[0][r]);
      pl[row * 32 + 16 + l15] = f32_bf16(s[1][r]);
    }
    asm volatile("s_wait_dscnt 0" ::: "memory");
    Frag16 ap;
    const unsigned* pl32 = (const unsigned*)pl;
#pragma unroll
    for (int j = 0; j < 8; ++j) ap.u[j] = pl32[(l15 * 32 + kA(j, half)) >> 1];
    // O += P @ V
#pragma unroll
    for (int dsub = 0; dsub < 4; ++dsub) {
      Frag16 bv;
      const int dcol = dsub * 16 + l15;
#pragma unroll
      for (int j = 0; j < 8; ++j) {
        int kk = kB(j, half);
#if HAVE_ASYNC
        bv.u[j] = vl32[(dcol * 32 + kk) >> 1];           // local 32-key row
#else
        bv.u[j] = ((const unsigned*)vraw)[(dcol * SEQ_PAD + key0 + kk) >> 1];
#endif
      }
      o[dsub] = __builtin_amdgcn_wmma_f32_16x16x32_bf16(
          false, ap.v, false, bv.v, (short)0, o[dsub], false, false);
    }
  }
  // normalize + store xo[b, n, h*64 + d] (f32)
  const int bi = bh / HEADS, h = bh % HEADS;
#pragma unroll
  for (int r = 0; r < 8; ++r) {
    int qr = q0 + r + 8 * half;
    if (qr < SEQ) {
      float inv = 1.0f / lrow[r];
      float* dst = xo + ((size_t)(bi * SEQ + qr)) * CH + h * HDIM;
#pragma unroll
      for (int dsub = 0; dsub < 4; ++dsub) dst[dsub * 16 + l15] = o[dsub][r] * inv;
    }
  }
}

// ---------------------------------------------------------------------------
// VQ: one wave per row. z = xo_row @ vq_in_w.T + b; argmin_e ||e||^2 - 2 z.e;
// store zq = codebook[idx]; accumulate sum((zq - z)^2) for loss.
// ---------------------------------------------------------------------------
__global__ __launch_bounds__(128) void vq_kernel(
    const float* __restrict__ xo, const float* __restrict__ win,
    const float* __restrict__ bin, const float* __restrict__ codebook,
    float* __restrict__ zq, float* __restrict__ loss_acc) {
  const int wave = blockIdx.x * 4 + (threadIdx.x >> 5);
  const int lane = threadIdx.x & 31;
  if (wave >= MROWS) return;
  const float* __restrict__ row = xo + (size_t)wave * CH;
  float z[4];
#pragma unroll
  for (int j = 0; j < 4; ++j) {
    float acc = 0.f;
    for (int c = lane; c < CH; c += 32) acc += row[c] * win[j * CH + c];
#pragma unroll
    for (int off = 16; off >= 1; off >>= 1) acc += __shfl_xor(acc, off, 32);
    z[j] = acc + bin[j];
  }
  float best = 1e30f;
  int bidx = 0x7fffffff;
  for (int e = lane; e < DICN; e += 32) {
    const float* cb = codebook + e * 4;
    float e0 = cb[0], e1 = cb[1], e2 = cb[2], e3 = cb[3];
    float d = e0 * e0 + e1 * e1 + e2 * e2 + e3 * e3 -
              2.f * (z[0] * e0 + z[1] * e1 + z[2] * e2 + z[3] * e3);
    if (d < best) { best = d; bidx = e; }
  }
#pragma unroll
  for (int off = 16; off >= 1; off >>= 1) {
    float ob = __shfl_xor(best, off, 32);
    int oi = __shfl_xor(bidx, off, 32);
    if (ob < best || (ob == best && oi < bidx)) { best = ob; bidx = oi; }
  }
  if (lane == 0) {
    float err = 0.f;
#pragma unroll
    for (int j = 0; j < 4; ++j) {
      float zqj = codebook[bidx * 4 + j];
      zq[(size_t)wave * 4 + j] = zqj;
      float dd = zqj - z[j];
      err += dd * dd;
    }
    atomicAdd(loss_acc, err);
  }
}

// ---------------------------------------------------------------------------
// Fused tail weights: Wf = proj_w @ vq_out_w (768x4), bf = proj_w@vq_out_b + proj_b
// ---------------------------------------------------------------------------
__global__ void wf_kernel(const float* __restrict__ proj_w,
                          const float* __restrict__ proj_b,
                          const float* __restrict__ vq_out_w,
                          const float* __restrict__ vq_out_b,
                          float* __restrict__ Wf, float* __restrict__ bf) {
  int c = blockIdx.x * blockDim.x + threadIdx.x;
  if (c >= CH) return;
  float a0 = 0.f, a1 = 0.f, a2 = 0.f, a3 = 0.f, ab = 0.f;
  for (int kx = 0; kx < CH; ++kx) {
    float pw = proj_w[(size_t)c * CH + kx];
    a0 += pw * vq_out_w[kx * 4 + 0];
    a1 += pw * vq_out_w[kx * 4 + 1];
    a2 += pw * vq_out_w[kx * 4 + 2];
    a3 += pw * vq_out_w[kx * 4 + 3];
    ab += pw * vq_out_b[kx];
  }
  Wf[c * 4 + 0] = a0; Wf[c * 4 + 1] = a1;
  Wf[c * 4 + 2] = a2; Wf[c * 4 + 3] = a3;
  bf[c] = ab + proj_b[c];
}

// out[m, c] = bf[c] + zq[m,:] . Wf[c,:]
__global__ void out_kernel(const float* __restrict__ zq,
                           const float* __restrict__ Wf,
                           const float* __restrict__ bfv,
                           float* __restrict__ out) {
  size_t i = (size_t)blockIdx.x * blockDim.x + threadIdx.x;
  if (i >= (size_t)MROWS * CH) return;
  int m = (int)(i / CH), c = (int)(i % CH);
  const float* zr = zq + (size_t)m * 4;
  const float* wr = Wf + (size_t)c * 4;
  out[i] = bfv[c] + zr[0] * wr[0] + zr[1] * wr[1] + zr[2] * wr[2] + zr[3] * wr[3];
}

__global__ void loss_kernel(const float* __restrict__ acc, float* __restrict__ out) {
  // loss = mean((sg(zq)-z)^2) + 0.25*mean((zq-sg(z))^2) = 1.25 * sum / (M*4)
  out[0] = 1.25f * acc[0] / ((float)MROWS * 4.0f);
}

extern "C" void kernel_launch(void* const* d_in, const int* in_sizes, int n_in,
                              void* d_out, int out_size, void* d_ws, size_t ws_size,
                              hipStream_t stream) {
  const float* x        = (const float*)d_in[0];
  const float* qkv_w    = (const float*)d_in[1];
  const float* proj_w   = (const float*)d_in[2];
  const float* proj_b   = (const float*)d_in[3];
  const float* vq_in_w  = (const float*)d_in[4];
  const float* vq_in_b  = (const float*)d_in[5];
  const float* vq_out_w = (const float*)d_in[6];
  const float* vq_out_b = (const float*)d_in[7];
  const float* codebook = (const float*)d_in[8];
  float* out = (float*)d_out;

  char* ws = (char*)d_ws;
  size_t off = 0;
  auto alloc = [&](size_t bytes) -> void* {
    void* p = ws + off;
    off = (off + bytes + 255) & ~(size_t)255;
    return p;
  };
  unsigned short* qb  = (unsigned short*)alloc((size_t)BATCH * HEADS * SEQ_PAD * HDIM * 2);
  unsigned short* kb  = (unsigned short*)alloc((size_t)BATCH * HEADS * SEQ_PAD * HDIM * 2);
  unsigned short* vtb = (unsigned short*)alloc((size_t)BATCH * HEADS * HDIM * SEQ_PAD * 2);
  float* xo  = (float*)alloc((size_t)MROWS * CH * 4);
  float* zq  = (float*)alloc((size_t)MROWS * 4 * 4);
  float* Wf  = (float*)alloc((size_t)CH * 4 * 4);
  float* bfv = (float*)alloc((size_t)CH * 4);
  float* loss_acc = (float*)alloc(256);

  // 1) zero pad rows + loss accumulator
  {
    int padN = BATCH * HEADS * (SEQ_PAD - SEQ) * HDIM;
    init_pad_kernel<<<(padN + 255) / 256, 256, 0, stream>>>(qb, kb, vtb, loss_acc);
  }
  // 2) QKV GEMM (WMMA bf16): 788*36 tile-waves (16x64 tiles), 4 waves/block
  qkv_wmma_kernel<<<(MT * NT4) / 4, 128, 0, stream>>>(x, qkv_w, qb, kb, vtb);
  // 3) Flash attention: 768 * 13 tile-waves
  attn_wmma_kernel<<<(BATCH * HEADS * QTILES) / 4, 128, 0, stream>>>(qb, kb, vtb, xo);
  // 4) VQ per row
  vq_kernel<<<(MROWS + 3) / 4, 128, 0, stream>>>(xo, vq_in_w, vq_in_b, codebook, zq, loss_acc);
  // 5) fused tail weights
  wf_kernel<<<(CH + 255) / 256, 256, 0, stream>>>(proj_w, proj_b, vq_out_w, vq_out_b, Wf, bfv);
  // 6) out = zq @ Wf.T + bf
  out_kernel<<<(int)(((size_t)MROWS * CH + 255) / 256), 256, 0, stream>>>(zq, Wf, bfv, out);
  // 7) scalar loss at the end of d_out
  loss_kernel<<<1, 1, 0, stream>>>(loss_acc, out + (size_t)MROWS * CH);
}
